// HAN_24300924961291
// MI455X (gfx1250) — compile-verified, matrix operand order checked by reading, not run on hardware
//
#include <hip/hip_runtime.h>
#include <hip/hip_bf16.h>

// ---------------- problem constants (match reference) ----------------
#define NA      50000      // N_AUTHOR
#define NP      100000     // N_PAPER
#define EREV    300000
#define ECO     200000
#define HH      8          // heads
#define DD      64         // head dim
#define DIMH    512        // H*D
#define DIMIN   128
#define DIMOUT  4
#define NEG_SLOPE 0.2f

typedef __bf16 bf16_t;
typedef __attribute__((ext_vector_type(16))) __bf16 v16bf;
typedef __attribute__((ext_vector_type(8)))  __bf16 v8bf;
typedef __attribute__((ext_vector_type(8)))  float  v8f;
typedef __attribute__((ext_vector_type(4))) unsigned int u32x4;
typedef __attribute__((ext_vector_type(8))) int i32x8;
typedef __attribute__((ext_vector_type(4))) int i32x4;

// ---------------- small utility kernels ----------------
__global__ void cvt_bf16_kernel(const float* __restrict__ src, bf16_t* __restrict__ dst, size_t n) {
    size_t i = (size_t)blockIdx.x * blockDim.x + threadIdx.x;
    if (i < n) dst[i] = (bf16_t)src[i];
}

__global__ void fill_kernel(float* __restrict__ p, float v, size_t n) {
    size_t i = (size_t)blockIdx.x * blockDim.x + threadIdx.x;
    if (i < n) p[i] = v;
}

__global__ void relu_kernel(float* __restrict__ p, size_t n) {
    size_t i = (size_t)blockIdx.x * blockDim.x + threadIdx.x;
    if (i < n) { float v = p[i]; p[i] = v > 0.f ? v : 0.f; }
}

// ---------------- Tensor Data Mover: 2-D tile -> LDS ----------------
// D# layout per CDNA5 ISA.
// data_size = 1 (2-byte elements). Group0: count=1, lds_addr, global_addr, type=2.
// Group1: data_size, tensor_dim0/1, tile_dim0/1, tensor_dim0_stride.
__device__ inline void tdm_load_2d(unsigned lds_off, const void* gptr,
                                   unsigned tensor_d0, unsigned tensor_d1,
                                   unsigned long long stride0_elems,
                                   unsigned tile_d0, unsigned tile_d1) {
    unsigned long long ga = (unsigned long long)(size_t)gptr;
    u32x4 g0 = {};
    g0[0] = 1u;                                                // count=1, user mode
    g0[1] = lds_off;                                           // lds_addr (bytes)   [63:32]
    g0[2] = (unsigned)(ga & 0xffffffffull);                    // global_addr[31:0]  [95:64]
    g0[3] = (unsigned)((ga >> 32) & 0x01ffffffull) | (2u << 30); // addr[56:32] | type=2
    i32x8 g1 = {};
    g1[0] = (int)(1u << 16);                                   // data_size=1 (2B)
    g1[1] = (int)((tensor_d0 & 0xffffu) << 16);                // tensor_dim0[15:0]  @bit48
    g1[2] = (int)(((tensor_d0 >> 16) & 0xffffu) | ((tensor_d1 & 0xffffu) << 16));
    g1[3] = (int)(((tensor_d1 >> 16) & 0xffffu) | ((tile_d0 & 0xffffu) << 16)); // tile_dim0 @bit112
    g1[4] = (int)(tile_d1 & 0xffffu);                          // tile_dim1 @bit128; tile_dim2=0
    g1[5] = (int)(unsigned)(stride0_elems & 0xffffffffull);    // tensor_dim0_stride[31:0] @bit160
    g1[6] = (int)(unsigned)((stride0_elems >> 32) & 0xffffull);// stride[47:32]; dim1_stride=0
    g1[7] = 0;
    i32x4 z4 = {};
#if __has_include(<hip/amd_detail/amd_gfx1250_TDM.h>)
    i32x8 z8 = {};
    __builtin_amdgcn_tensor_load_to_lds(g0, g1, z4, z4, z8, 0);   // 6-arg toolchain
#else
    __builtin_amdgcn_tensor_load_to_lds(g0, g1, z4, z4, 0);       // 5-arg (ROCm 7.2)
#endif
}

// ---------------- WMMA bf16 GEMM core (TDM-staged) ----------------
// Block = 256 threads = 8 waves, computing a 16(M) x 128(N) super-tile.
// Per K-step(32): wave0 issues two TDM loads (A 16x32 tile, B 32x128 panel) into
// LDS, waits TENSORcnt, barrier; each wave then feeds one
// v_wmma_f32_16x16x32_bf16 from LDS. Register striping per ISA 7.12.2.
__device__ inline v8f gemm_block_core(const bf16_t* __restrict__ A, const bf16_t* __restrict__ B,
                                      int M, int N, int K, int tm, int tnB,
                                      bf16_t* sA, bf16_t* sB) {
    const int w    = threadIdx.x >> 5;
    const int lane = threadIdx.x & 31;
    const int r    = lane & 15;
    const int hi   = lane >> 4;
    v8f acc = {};
    for (int k0 = 0; k0 < K; k0 += 32) {
        __syncthreads();                                  // prior-step LDS reads done
        if (w == 0) {
            // A tile: rows [tm*16, +16), cols [k0, +32) of A[M][K]
            tdm_load_2d((unsigned)(size_t)sA,
                        A + (size_t)(tm * 16) * K + k0,
                        (unsigned)K, (unsigned)M, (unsigned long long)K, 32u, 16u);
            // B panel: rows [k0, +32), cols [tnB*128, +128) of B[K][N]
            tdm_load_2d((unsigned)(size_t)sB,
                        B + (size_t)k0 * N + (size_t)tnB * 128,
                        (unsigned)N, (unsigned)K, (unsigned long long)N, 128u, 32u);
            __builtin_amdgcn_s_wait_tensorcnt(0);
        }
        __syncthreads();                                  // publish LDS to all waves

        // A operand: lane holds row r, K-runs {8*hi..8*hi+7} and {16+8*hi..23+8*hi}
        v8bf alo = *(const v8bf*)(sA + r * 32 + 8 * hi);
        v8bf ahi = *(const v8bf*)(sA + r * 32 + 16 + 8 * hi);
        v16bf a, b;
#pragma unroll
        for (int i = 0; i < 8; ++i) { a[i] = alo[i]; a[8 + i] = ahi[i]; }
        // B operand: lane holds column (w*16 + r), same K striping
        const int col = w * 16 + r;
#pragma unroll
        for (int i = 0; i < 8; ++i) {
            b[i]     = sB[(8 * hi + i) * 128 + col];
            b[8 + i] = sB[(16 + 8 * hi + i) * 128 + col];
        }
        acc = __builtin_amdgcn_wmma_f32_16x16x32_bf16(false, a, false, b, (short)0, acc,
                                                      false, false);
    }
    return acc;
}

// C = A @ B + bias  (store f32)
__global__ void gemm_bias_kernel(const bf16_t* __restrict__ A, const bf16_t* __restrict__ B,
                                 const float* __restrict__ bias, float* __restrict__ C,
                                 int M, int N, int K) {
    __shared__ bf16_t sA[16 * 32];     // 1 KB
    __shared__ bf16_t sB[32 * 128];    // 8 KB
    const int nPanels = N >> 7;        // N/128
    const int tm  = (int)blockIdx.x / nPanels;
    const int tnB = (int)blockIdx.x % nPanels;
    v8f acc = gemm_block_core(A, B, M, N, K, tm, tnB, sA, sB);
    const int w = threadIdx.x >> 5, lane = threadIdx.x & 31;
    const int r = lane & 15, hi = lane >> 4;
    const int col = tnB * 128 + w * 16 + r;
    const float bv = bias[col];
#pragma unroll
    for (int i = 0; i < 8; ++i) {
        const int row = tm * 16 + i + 8 * hi;
        C[(size_t)row * (size_t)N + col] = acc[i] + bv;
    }
}

// fused semantic attention logit:  atomicAdd(wout, sum_tile tanh(A@B + k_b) * q)
__global__ void gemm_tanh_reduce_kernel(const bf16_t* __restrict__ A, const bf16_t* __restrict__ B,
                                        const float* __restrict__ kb, const float* __restrict__ q,
                                        float* __restrict__ wout, int M, int N, int K) {
    __shared__ bf16_t sA[16 * 32];
    __shared__ bf16_t sB[32 * 128];
    const int nPanels = N >> 7;
    const int tm  = (int)blockIdx.x / nPanels;
    const int tnB = (int)blockIdx.x % nPanels;
    v8f acc = gemm_block_core(A, B, M, N, K, tm, tnB, sA, sB);
    const int w = threadIdx.x >> 5, lane = threadIdx.x & 31;
    const int r = lane & 15;
    const int col = tnB * 128 + w * 16 + r;
    const float kbv = kb[col], qv = q[col];
    float part = 0.f;
#pragma unroll
    for (int i = 0; i < 8; ++i) part += tanhf(acc[i] + kbv) * qv;
#pragma unroll
    for (int off = 16; off > 0; off >>= 1) part += __shfl_xor(part, off, 32);
    if (lane == 0) atomicAdd(wout, part);
}

// ---------------- GAT edge-softmax kernels (fp32, atomics) ----------------
__device__ inline void atomicMaxFloat(float* addr, float val) {
    if (val >= 0.f) atomicMax((int*)addr, __float_as_int(val));
    else            atomicMin((unsigned int*)addr, __float_as_uint(val));
}

// per-node per-head alpha contribution: out[n,h] = sum_d h[n,h*64+d] * att[h*64+d]
__global__ void node_alpha_kernel(const float* __restrict__ h, const float* __restrict__ att,
                                  float* __restrict__ out, int n_nodes) {
    int t = blockIdx.x * blockDim.x + threadIdx.x;
    if (t >= n_nodes * HH) return;
    int n = t >> 3, hh = t & 7;
    const float* hp = h + (size_t)n * DIMH + hh * DD;
    const float* ap = att + hh * DD;
    float s = 0.f;
#pragma unroll 8
    for (int d = 0; d < DD; ++d) s += hp[d] * ap[d];
    out[t] = s;
}

__global__ void edge_max_kernel(const float* __restrict__ as_, const float* __restrict__ ad_,
                                const int* __restrict__ src, const int* __restrict__ dst,
                                float* __restrict__ al, float* __restrict__ m, int E) {
    int t = blockIdx.x * blockDim.x + threadIdx.x;
    if (t >= E * HH) return;
    int e = t >> 3, h = t & 7;
    int s = src[e], d = dst[e];
    float v = as_[s * HH + h] + ad_[d * HH + h];
    v = (v > 0.f) ? v : NEG_SLOPE * v;          // leaky_relu
    al[t] = v;
    atomicMaxFloat(&m[d * HH + h], v);
}

__global__ void edge_expsum_kernel(float* __restrict__ al, const float* __restrict__ m,
                                   float* __restrict__ s, const int* __restrict__ dst, int E) {
    int t = blockIdx.x * blockDim.x + threadIdx.x;
    if (t >= E * HH) return;
    int e = t >> 3, h = t & 7;
    int d = dst[e];
    float ex = expf(al[t] - m[d * HH + h]);
    al[t] = ex;                                  // reuse buffer: now holds e
    atomicAdd(&s[d * HH + h], ex);
}

__global__ void edge_scatter_kernel(const float* __restrict__ al, const float* __restrict__ s,
                                    const float* __restrict__ hsrc, const int* __restrict__ src,
                                    const int* __restrict__ dst, float* __restrict__ out,
                                    long long total) {
    long long t = (long long)blockIdx.x * blockDim.x + threadIdx.x;
    if (t >= total) return;
    int e = (int)(t >> 9);
    int f = (int)(t & 511);
    int h = f >> 6;
    int sn = src[e], d = dst[e];
    float a = al[e * HH + h] / s[(size_t)d * HH + h];
    atomicAdd(&out[(size_t)d * DIMH + f], a * hsrc[(size_t)sn * DIMH + f]);
}

// ---------------- final combine: beta-softmax + 512x4 GEMV ----------------
__global__ void final_kernel(const float* __restrict__ rev, const float* __restrict__ co,
                             const float* __restrict__ w, const float* __restrict__ linW,
                             const float* __restrict__ linb, float* __restrict__ out, int n_nodes) {
    const int wave = (int)((blockIdx.x * blockDim.x + threadIdx.x) >> 5);
    const int lane = threadIdx.x & 31;
    if (wave >= n_nodes) return;
    const float w0 = w[0] * (1.f / (float)NA);
    const float w1 = w[1] * (1.f / (float)NA);
    const float mx = fmaxf(w0, w1);
    const float e0 = expf(w0 - mx), e1 = expf(w1 - mx);
    const float inv = 1.f / (e0 + e1);
    const float b0 = e0 * inv, b1 = e1 * inv;
    const float* r = rev + (size_t)wave * DIMH;
    const float* c = co  + (size_t)wave * DIMH;
    float a0 = 0.f, a1 = 0.f, a2 = 0.f, a3 = 0.f;
#pragma unroll
    for (int i = 0; i < 16; ++i) {
        const int f = lane + i * 32;
        const float v = b0 * r[f] + b1 * c[f];
        const float* lw = linW + f * 4;
        a0 += v * lw[0]; a1 += v * lw[1]; a2 += v * lw[2]; a3 += v * lw[3];
    }
#pragma unroll
    for (int off = 16; off > 0; off >>= 1) {
        a0 += __shfl_xor(a0, off, 32); a1 += __shfl_xor(a1, off, 32);
        a2 += __shfl_xor(a2, off, 32); a3 += __shfl_xor(a3, off, 32);
    }
    if (lane == 0) {
        out[wave * 4 + 0] = a0 + linb[0];
        out[wave * 4 + 1] = a1 + linb[1];
        out[wave * 4 + 2] = a2 + linb[2];
        out[wave * 4 + 3] = a3 + linb[3];
    }
}

// ---------------- host orchestration ----------------
static inline int cdiv_ll(long long a, long long b) { return (int)((a + b - 1) / b); }

extern "C" void kernel_launch(void* const* d_in, const int* in_sizes, int n_in,
                              void* d_out, int out_size, void* d_ws, size_t ws_size,
                              hipStream_t stream) {
    (void)in_sizes; (void)n_in; (void)out_size; (void)ws_size;
    const float* x_a   = (const float*)d_in[0];
    const float* x_p   = (const float*)d_in[1];
    const float* W_a   = (const float*)d_in[2];
    const float* b_a   = (const float*)d_in[3];
    const float* W_p   = (const float*)d_in[4];
    const float* b_p   = (const float*)d_in[5];
    const float* att_rs = (const float*)d_in[6];
    const float* att_rd = (const float*)d_in[7];
    const float* att_cs = (const float*)d_in[8];
    const float* att_cd = (const float*)d_in[9];
    const float* k_W   = (const float*)d_in[10];
    const float* k_b   = (const float*)d_in[11];
    const float* q     = (const float*)d_in[12];
    const float* lin_W = (const float*)d_in[13];
    const float* lin_b = (const float*)d_in[14];
    const int* rev_src = (const int*)d_in[15];
    const int* rev_dst = (const int*)d_in[16];
    const int* co_src  = (const int*)d_in[17];
    const int* co_dst  = (const int*)d_in[18];
    float* out = (float*)d_out;

    // ---- workspace carve (persistent region + phase-reused pool) ----
    char* ws = (char*)d_ws;
    size_t off = 0;
    auto take = [&](size_t bytes) -> char* {
        char* p = ws + off; off += (bytes + 255) & ~(size_t)255; return p;
    };
    float* h_a     = (float*)take((size_t)NA * DIMH * 4);   // author features
    float* h_p     = (float*)take((size_t)NP * DIMH * 4);   // paper features
    float* out_rev = (float*)take((size_t)NA * DIMH * 4);
    float* out_co  = (float*)take((size_t)NA * DIMH * 4);
    char* pool = ws + off;

    // Phase A layout (bf16 conversions for projection GEMMs)
    bf16_t* xA_bf = (bf16_t*)pool;
    bf16_t* xP_bf = xA_bf + (size_t)NA * DIMIN;
    bf16_t* WA_bf = xP_bf + (size_t)NP * DIMIN;
    bf16_t* WP_bf = WA_bf + (size_t)DIMIN * DIMH;

    // Phase B layout (edge softmax buffers) — reuses the same pool
    float* as_rev = (float*)pool;
    float* ad_rev = as_rev + (size_t)NP * HH;
    float* as_co  = ad_rev + (size_t)NA * HH;
    float* ad_co  = as_co  + (size_t)NA * HH;
    float* m_rev  = ad_co  + (size_t)NA * HH;
    float* s_rev  = m_rev  + (size_t)NA * HH;
    float* m_co   = s_rev  + (size_t)NA * HH;
    float* s_co   = m_co   + (size_t)NA * HH;
    float* al_rev = s_co   + (size_t)NA * HH;
    float* al_co  = al_rev + (size_t)EREV * HH;

    // Phase C layout (semantic attention) — overlays h_a / h_p (dead by then)
    bf16_t* kW_bf  = (bf16_t*)h_a;
    float*  w2     = (float*)((char*)h_a + (1 << 20));
    bf16_t* rev_bf = (bf16_t*)h_p;
    bf16_t* co_bf  = rev_bf + (size_t)NA * DIMH;

    const int B = 256;
    const float NEG_INF = -__builtin_huge_valf();

    // ---- Phase A: projections via TDM-staged WMMA ----
    cvt_bf16_kernel<<<cdiv_ll((long long)NA * DIMIN, B), B, 0, stream>>>(x_a, xA_bf, (size_t)NA * DIMIN);
    cvt_bf16_kernel<<<cdiv_ll((long long)NP * DIMIN, B), B, 0, stream>>>(x_p, xP_bf, (size_t)NP * DIMIN);
    cvt_bf16_kernel<<<cdiv_ll((long long)DIMIN * DIMH, B), B, 0, stream>>>(W_a, WA_bf, (size_t)DIMIN * DIMH);
    cvt_bf16_kernel<<<cdiv_ll((long long)DIMIN * DIMH, B), B, 0, stream>>>(W_p, WP_bf, (size_t)DIMIN * DIMH);
    gemm_bias_kernel<<<(NA / 16) * (DIMH / 128), B, 0, stream>>>(xA_bf, WA_bf, b_a, h_a, NA, DIMH, DIMIN);
    gemm_bias_kernel<<<(NP / 16) * (DIMH / 128), B, 0, stream>>>(xP_bf, WP_bf, b_p, h_p, NP, DIMH, DIMIN);

    // ---- Phase B: GAT edge attention (scatter softmax) ----
    node_alpha_kernel<<<cdiv_ll((long long)NP * HH, B), B, 0, stream>>>(h_p, att_rs, as_rev, NP);
    node_alpha_kernel<<<cdiv_ll((long long)NA * HH, B), B, 0, stream>>>(h_a, att_rd, ad_rev, NA);
    node_alpha_kernel<<<cdiv_ll((long long)NA * HH, B), B, 0, stream>>>(h_a, att_cs, as_co, NA);
    node_alpha_kernel<<<cdiv_ll((long long)NA * HH, B), B, 0, stream>>>(h_a, att_cd, ad_co, NA);

    fill_kernel<<<cdiv_ll((long long)NA * HH, B), B, 0, stream>>>(m_rev, NEG_INF, (size_t)NA * HH);
    fill_kernel<<<cdiv_ll((long long)NA * HH, B), B, 0, stream>>>(s_rev, 0.f, (size_t)NA * HH);
    fill_kernel<<<cdiv_ll((long long)NA * HH, B), B, 0, stream>>>(m_co, NEG_INF, (size_t)NA * HH);
    fill_kernel<<<cdiv_ll((long long)NA * HH, B), B, 0, stream>>>(s_co, 0.f, (size_t)NA * HH);
    fill_kernel<<<cdiv_ll((long long)NA * DIMH, B), B, 0, stream>>>(out_rev, 0.f, (size_t)NA * DIMH);
    fill_kernel<<<cdiv_ll((long long)NA * DIMH, B), B, 0, stream>>>(out_co, 0.f, (size_t)NA * DIMH);

    edge_max_kernel<<<cdiv_ll((long long)EREV * HH, B), B, 0, stream>>>(as_rev, ad_rev, rev_src, rev_dst, al_rev, m_rev, EREV);
    edge_expsum_kernel<<<cdiv_ll((long long)EREV * HH, B), B, 0, stream>>>(al_rev, m_rev, s_rev, rev_dst, EREV);
    edge_scatter_kernel<<<cdiv_ll((long long)EREV * DIMH, B), B, 0, stream>>>(al_rev, s_rev, h_p, rev_src, rev_dst, out_rev, (long long)EREV * DIMH);

    edge_max_kernel<<<cdiv_ll((long long)ECO * HH, B), B, 0, stream>>>(as_co, ad_co, co_src, co_dst, al_co, m_co, ECO);
    edge_expsum_kernel<<<cdiv_ll((long long)ECO * HH, B), B, 0, stream>>>(al_co, m_co, s_co, co_dst, ECO);
    edge_scatter_kernel<<<cdiv_ll((long long)ECO * DIMH, B), B, 0, stream>>>(al_co, s_co, h_a, co_src, co_dst, out_co, (long long)ECO * DIMH);

    relu_kernel<<<cdiv_ll((long long)NA * DIMH, B), B, 0, stream>>>(out_rev, (size_t)NA * DIMH);
    relu_kernel<<<cdiv_ll((long long)NA * DIMH, B), B, 0, stream>>>(out_co, (size_t)NA * DIMH);

    // ---- Phase C: semantic attention (fused TDM+WMMA + tanh reduction) ----
    cvt_bf16_kernel<<<cdiv_ll((long long)NA * DIMH, B), B, 0, stream>>>(out_rev, rev_bf, (size_t)NA * DIMH);
    cvt_bf16_kernel<<<cdiv_ll((long long)NA * DIMH, B), B, 0, stream>>>(out_co, co_bf, (size_t)NA * DIMH);
    cvt_bf16_kernel<<<cdiv_ll((long long)DIMH * DIMH, B), B, 0, stream>>>(k_W, kW_bf, (size_t)DIMH * DIMH);
    fill_kernel<<<1, B, 0, stream>>>(w2, 0.f, 2);
    gemm_tanh_reduce_kernel<<<(NA / 16) * (DIMH / 128), B, 0, stream>>>(rev_bf, kW_bf, k_b, q, &w2[0], NA, DIMH, DIMH);
    gemm_tanh_reduce_kernel<<<(NA / 16) * (DIMH / 128), B, 0, stream>>>(co_bf, kW_bf, k_b, q, &w2[1], NA, DIMH, DIMH);

    // ---- final: beta softmax + [512x4] GEMV ----
    final_kernel<<<cdiv_ll((long long)NA * 32, B), B, 0, stream>>>(out_rev, out_co, w2, lin_W, lin_b, out, NA);
}
